// gcnlayer_75771813036287
// MI455X (gfx1250) — matrix operand order repaired and checked by reading, not compiled, not run on hardware
//
#include <hip/hip_runtime.h>
#include <hip/hip_bf16.h>
#include <cmath>

typedef __bf16 bf16_t;
typedef __attribute__((ext_vector_type(16))) __bf16 v16bf;
typedef __attribute__((ext_vector_type(8)))  float  v8f;

#define N_NODES 12000
#define NF      64
#define NDIM    64
#define KTILE   32
#define NKT     (N_NODES / KTILE)   // 375
#define MTILE   16
#define NWAVES  4

// ---------------------------------------------------------------------------
// Kernel 1: X [N,64] f32 (row-major)  ->  XbT [64,N] bf16 (transposed)
// Makes WMMA B-fragment loads contiguous 32B per lane.
// ---------------------------------------------------------------------------
__global__ void cvt_transpose_bf16(const float* __restrict__ X,
                                   bf16_t* __restrict__ XbT) {
  int idx = blockIdx.x * blockDim.x + threadIdx.x;
  if (idx >= N_NODES * NF) return;
  int j = idx >> 6;       // node
  int f = idx & 63;       // feature
  XbT[(size_t)f * N_NODES + j] = (bf16_t)X[(size_t)j * NF + f];
}

// ---------------------------------------------------------------------------
// Kernel 2: fused  out = tanh( (A * P/EP) @ X @ Wd + b )
// One workgroup = one 16-row tile; 4 waves split the K loop.
// ---------------------------------------------------------------------------
__global__ __launch_bounds__(128)
void gcn_fused(const int*    __restrict__ A,
               const float*  __restrict__ P,
               const float*  __restrict__ EP,
               const float*  __restrict__ Wd,
               const float*  __restrict__ bvec,
               const bf16_t* __restrict__ XbT,
               float*        __restrict__ out) {
  __shared__ float part[NWAVES * MTILE * NF];  // per-wave partial agg (16 KB)
  __shared__ float agg [MTILE * NF];           // reduced, P-scaled agg (4 KB)

  const int tid     = threadIdx.x;
  const int wave    = tid >> 5;
  const int lane    = tid & 31;
  const int ln      = lane & 15;
  const int hi      = lane >> 4;       // lane half (0 or 1)
  const int hi8     = hi * 8;
  const int rowBase = blockIdx.x * MTILE;
  const int row     = rowBase + ln;    // lanes L and L+16 share row M=L

  const int*   Arow  = A  + (size_t)row * N_NODES;
  const float* EProw = EP + (size_t)row * N_NODES;

  // Per-lane B row pointers: column n = j*16 + ln of the big GEMM,
  // lane half selects K offset 0 / 16 within a K-tile.
  const bf16_t* Brow[4];
#pragma unroll
  for (int j = 0; j < 4; ++j)
    Brow[j] = XbT + (size_t)(j * 16 + ln) * N_NODES + hi * 16;

  v8f acc[4];
#pragma unroll
  for (int j = 0; j < 4; ++j) acc[j] = (v8f)0.0f;

  // --- K loop: 375 K-tiles of 32, strided across the 4 waves -------------
  for (int kt = wave; kt < NKT; kt += NWAVES) {
    const int kb = kt * KTILE;
    const int k1 = kb + hi * 16;    // A-frag elems 0..7  : K = k1..k1+7
    const int k2 = k1 + 8;         // A-frag elems 8..15 : K = k2..k2+7

    int4   a0 = *(const int4*)  (Arow  + k1);
    int4   a1 = *(const int4*)  (Arow  + k1 + 4);
    int4   a2 = *(const int4*)  (Arow  + k2);
    int4   a3 = *(const int4*)  (Arow  + k2 + 4);
    float4 e0 = *(const float4*)(EProw + k1);
    float4 e1 = *(const float4*)(EProw + k1 + 4);
    float4 e2 = *(const float4*)(EProw + k2);
    float4 e3 = *(const float4*)(EProw + k2 + 4);

    // speculative prefetch of this wave's next K-tile (global_prefetch_b8)
    __builtin_prefetch(Arow  + k1 + NWAVES * KTILE, 0, 0);
    __builtin_prefetch(EProw + k1 + NWAVES * KTILE, 0, 0);

    // w' = A ? 1/EP : 0   (P factored out; applied after reduction)
    v16bf af;
#define WELT(e, ai, ei) af[e] = (ai) ? (bf16_t)__builtin_amdgcn_rcpf(ei) : (bf16_t)0.0f
    WELT(0,  a0.x, e0.x); WELT(1,  a0.y, e0.y); WELT(2,  a0.z, e0.z); WELT(3,  a0.w, e0.w);
    WELT(4,  a1.x, e1.x); WELT(5,  a1.y, e1.y); WELT(6,  a1.z, e1.z); WELT(7,  a1.w, e1.w);
    WELT(8,  a2.x, e2.x); WELT(9,  a2.y, e2.y); WELT(10, a2.z, e2.z); WELT(11, a2.w, e2.w);
    WELT(12, a3.x, e3.x); WELT(13, a3.y, e3.y); WELT(14, a3.z, e3.z); WELT(15, a3.w, e3.w);
#undef WELT

#pragma unroll
    for (int j = 0; j < 4; ++j) {
      v16bf bf = *(const v16bf*)(Brow[j] + kb);
      acc[j] = __builtin_amdgcn_wmma_f32_16x16x32_bf16(
          /*neg_a=*/false, af, /*neg_b=*/false, bf,
          /*c_mod=*/(short)0, acc[j], /*reuse_a=*/false, /*reuse_b=*/false);
    }
  }

  // --- stash this wave's 16x64 partial to LDS ----------------------------
  // C layout: VGPR r -> M = r + hi*8 (lanes 0-15 / 16-31), N = lane&15
  float* my = part + wave * (MTILE * NF);
#pragma unroll
  for (int j = 0; j < 4; ++j)
#pragma unroll
    for (int r = 0; r < 8; ++r)
      my[(r + hi8) * NF + j * 16 + ln] = acc[j][r];

  __syncthreads();

  // --- reduce 4 partials, scale by P[i] ----------------------------------
  const int m  = tid >> 3;          // 0..15
  const int c0 = (tid & 7) * 8;     // 0,8,..,56
  const float pv = P[rowBase + m];
#pragma unroll
  for (int c = 0; c < 8; ++c) {
    int ix = m * NF + c0 + c;
    float s = part[ix] + part[MTILE * NF + ix] +
              part[2 * MTILE * NF + ix] + part[3 * MTILE * NF + ix];
    agg[ix] = s * pv;
  }
  __syncthreads();

  // --- tiny second GEMM + tanh: out[m, c0..c0+7] -------------------------
  float accf[8];
#pragma unroll
  for (int j = 0; j < 8; ++j) accf[j] = bvec[c0 + j];

  for (int f = 0; f < NF; ++f) {
    float  av = agg[m * NF + f];
    float4 w0 = *(const float4*)(Wd + f * NDIM + c0);
    float4 w1 = *(const float4*)(Wd + f * NDIM + c0 + 4);
    accf[0] += av * w0.x; accf[1] += av * w0.y;
    accf[2] += av * w0.z; accf[3] += av * w0.w;
    accf[4] += av * w1.x; accf[5] += av * w1.y;
    accf[6] += av * w1.z; accf[7] += av * w1.w;
  }

  float* orow = out + (size_t)(rowBase + m) * NDIM + c0;
#pragma unroll
  for (int j = 0; j < 8; ++j) orow[j] = tanhf(accf[j]);
}

// ---------------------------------------------------------------------------
extern "C" void kernel_launch(void* const* d_in, const int* in_sizes, int n_in,
                              void* d_out, int out_size, void* d_ws, size_t ws_size,
                              hipStream_t stream) {
  const float* X  = (const float*)d_in[0];   // [N,64] f32
  const int*   A  = (const int*)  d_in[1];   // [N,N]  i32
  const float* P  = (const float*)d_in[2];   // [N]    f32
  const float* EP = (const float*)d_in[3];   // [N,N]  f32
  const float* Wd = (const float*)d_in[4];   // [64,64] f32
  const float* b  = (const float*)d_in[5];   // [64]   f32
  float* out  = (float*)d_out;               // [N,64] f32
  bf16_t* XbT = (bf16_t*)d_ws;               // [64,N] bf16 (1.5 MB scratch)

  cvt_transpose_bf16<<<(N_NODES * NF + 255) / 256, 256, 0, stream>>>(X, XbT);
  gcn_fused<<<N_NODES / MTILE, 128, 0, stream>>>(A, P, EP, Wd, b, XbT, out);
}
